// Attention_59493886984703
// MI455X (gfx1250) — matrix operand order repaired and checked by reading, not compiled, not run on hardware
//
#include <hip/hip_runtime.h>

// ---------------------------------------------------------------------------
// Linear attention pipeline for MI455X (gfx1250, wave32, WMMA 16x16x32 f16).
// Compute-bound (~283 GFLOP vs ~0.5 GB traffic) => all GEMMs through
// v_wmma_f32_16x16x32_f16. Fragment reads are contiguous ds_load_b128;
// operand transposes use single v_perm_b32 packs at tile-load time; A-tiles
// are staged with GLOBAL_LOAD_ASYNC_TO_LDS_B128 (ASYNCcnt) when available;
// LDS is double-buffered (one barrier per K-chunk).
// ---------------------------------------------------------------------------

typedef __attribute__((ext_vector_type(16))) _Float16 v16h;
typedef __attribute__((ext_vector_type(8)))  float    v8f;
typedef __attribute__((ext_vector_type(4)))  int      v4i_;
typedef _Float16 half_t;

#define BQ   8
#define SQ   4096
#define DQ   1024
#define HQ   16
#define DHQ  64
#define DMQ  1024
#define BSQ  (BQ * SQ)          // 32768 rows

#if defined(__AMDGCN__) && __has_builtin(__builtin_amdgcn_global_load_async_to_lds_b128)
#define HAVE_ASYNC_LDS 1
#else
#define HAVE_ASYNC_LDS 0
#endif

struct Blk32B { uint4 lo, hi; };

// load 16 contiguous halves (32B) from LDS -> v16h fragment
static __device__ inline v16h frag_contig(const half_t* p) {
  Blk32B t;
  t.lo = *(const uint4*)p;
  t.hi = *(const uint4*)(p + 8);
  return __builtin_bit_cast(v16h, t);
}

// 16B global -> LDS copy; async DMA path on gfx1250 toolchains that have it
static __device__ inline void copy16_g2l(half_t* lds_dst, const half_t* gsrc) {
#if HAVE_ASYNC_LDS
  __builtin_amdgcn_global_load_async_to_lds_b128(
      (__attribute__((address_space(1))) v4i_*)(unsigned long long)gsrc,
      (__attribute__((address_space(3))) v4i_*)(unsigned)(unsigned long long)lds_dst,
      0, 0);
#else
  *(uint4*)lds_dst = *(const uint4*)gsrc;
#endif
}

static __device__ inline void async_fence() {
#if HAVE_ASYNC_LDS
#if __has_builtin(__builtin_amdgcn_s_wait_asynccnt)
  __builtin_amdgcn_s_wait_asynccnt(0);
#else
  asm volatile("s_wait_asynccnt 0x0" ::: "memory");
#endif
#endif
}

// merge low/high halves of two k-rows into one dword: (r0.h | r1.h << 16)
static __device__ inline uint perm_lo(uint r0d, uint r1d) {
  return __builtin_amdgcn_perm(r1d, r0d, 0x05040100u);
}
static __device__ inline uint perm_hi(uint r0d, uint r1d) {
  return __builtin_amdgcn_perm(r1d, r0d, 0x07060302u);
}

// ------------------------------ f32 -> f16 ---------------------------------
__global__ __launch_bounds__(256) void f32_to_f16_kernel(
    const float* __restrict__ in, half_t* __restrict__ out, long n) {
  const long i = ((long)blockIdx.x * 256 + threadIdx.x) * 8;
  if (i >= n) return;
  const float4 a = *(const float4*)(in + i);
  const float4 b = *(const float4*)(in + i + 4);
  union { uint4 u; half_t h[8]; } o;
  o.h[0] = (half_t)a.x; o.h[1] = (half_t)a.y;
  o.h[2] = (half_t)a.z; o.h[3] = (half_t)a.w;
  o.h[4] = (half_t)b.x; o.h[5] = (half_t)b.y;
  o.h[6] = (half_t)b.z; o.h[7] = (half_t)b.w;
  *(uint4*)(out + i) = o.u;
}

// ------------------------- generic f16 WMMA GEMM ---------------------------
// C(M,N) = A(M,K) @ B(K,N) [+ bias]; A,B row-major f16.
// 128 threads = 4 waves; block tile 128x64, wave tile 32x64, K chunk 32.
// sA: row-major [m][k] (async-staged). sB: column-major [n][k] (perm scatter).
// Double-buffered: one barrier per K-chunk.
__global__ __launch_bounds__(128) void gemm_f16_kernel(
    const half_t* __restrict__ A, const half_t* __restrict__ B,
    const float* __restrict__ bias,
    half_t* __restrict__ Ch, float* __restrict__ Cf,
    int M, int N, int K) {
  constexpr int BK = 32, LDA = BK + 8, LDB = BK + 8;  // even strides (halves)
  __shared__ half_t sA[2][128 * LDA];
  __shared__ half_t sB[2][64 * LDB];

  const int tid  = threadIdx.x;
  const int lane = tid & 31;
  const int wave = tid >> 5;
  const int m0 = blockIdx.y * 128;
  const int n0 = blockIdx.x * 64;

  const int brp = tid >> 3, bn0 = (tid & 7) * 8;  // B: 16 k-pairs x 8 cols

  auto stage = [&](int buf, int k0) {
    // A tile 128x32: one row (2x16B async DMA) per thread
    const half_t* gA = A + (long)(m0 + tid) * K + k0;
    copy16_g2l(sA[buf] + tid * LDA,      gA);
    copy16_g2l(sA[buf] + tid * LDA + 8,  gA + 8);
    // B tile 32x64 -> column-major [n][k], k-pairs packed via v_perm_b32
    const half_t* gB = B + (long)(k0 + 2 * brp) * N + n0 + bn0;
    uint4 r0 = *(const uint4*)gB;
    uint4 r1 = *(const uint4*)(gB + N);
    const uint* d0 = (const uint*)&r0;
    const uint* d1 = (const uint*)&r1;
#pragma unroll
    for (int w = 0; w < 4; ++w) {
      *(uint*)(sB[buf] + (bn0 + 2 * w)     * LDB + 2 * brp) = perm_lo(d0[w], d1[w]);
      *(uint*)(sB[buf] + (bn0 + 2 * w + 1) * LDB + 2 * brp) = perm_hi(d0[w], d1[w]);
    }
  };

  v8f acc[2][4] = {};

  stage(0, 0);
  async_fence();
  __syncthreads();

  const int nIter = K / BK;
  for (int it = 0; it < nIter; ++it) {
    const int cur = it & 1;
    if (it + 1 < nIter) stage(cur ^ 1, (it + 1) * BK);

    const half_t* bA = sA[cur];
    const half_t* bB = sB[cur];
    v16h afrag[2];
#pragma unroll
    for (int mt = 0; mt < 2; ++mt)
      afrag[mt] = frag_contig(
          bA + (wave * 32 + mt * 16 + (lane & 15)) * LDA + ((lane >> 4) << 4));
#pragma unroll
    for (int nt = 0; nt < 4; ++nt) {
      const v16h bfrag =
          frag_contig(bB + (nt * 16 + (lane & 15)) * LDB + ((lane >> 4) << 4));
#pragma unroll
      for (int mt = 0; mt < 2; ++mt)
        acc[mt][nt] = __builtin_amdgcn_wmma_f32_16x16x32_f16(
            false, afrag[mt], false, bfrag, (short)0, acc[mt][nt], false, false);
    }
    async_fence();
    __syncthreads();
  }

#pragma unroll
  for (int mt = 0; mt < 2; ++mt)
#pragma unroll
    for (int nt = 0; nt < 4; ++nt) {
      const int n = n0 + nt * 16 + (lane & 15);
      const float bv = bias ? bias[n] : 0.0f;
#pragma unroll
      for (int r = 0; r < 8; ++r) {
        const int m = m0 + wave * 32 + mt * 16 + r + ((lane >> 4) << 3);
        const float v = acc[mt][nt][r] + bv;
        if (Cf) Cf[(long)m * N + n] = v;
        else    Ch[(long)m * N + n] = (half_t)v;
      }
    }
}

// ----------------- softmax over last dim (DH=64), one wave/group -----------
__global__ __launch_bounds__(256) void softmax_q_kernel(
    const half_t* __restrict__ qin, half_t* __restrict__ qout, float scale) {
  const long g    = (long)blockIdx.x * 8 + (threadIdx.x >> 5);
  const int  lane = threadIdx.x & 31;
  union { uint u; half_t h[2]; } p;
  p.u = *(const uint*)(qin + g * 64 + lane * 2);
  const float v0 = (float)p.h[0], v1 = (float)p.h[1];
  float mx = fmaxf(v0, v1);
#pragma unroll
  for (int off = 16; off > 0; off >>= 1) mx = fmaxf(mx, __shfl_xor(mx, off, 32));
  const float e0 = __expf(v0 - mx), e1 = __expf(v1 - mx);
  float s = e0 + e1;
#pragma unroll
  for (int off = 16; off > 0; off >>= 1) s += __shfl_xor(s, off, 32);
  const float inv = scale / s;
  union { uint u; half_t h[2]; } o;
  o.h[0] = (half_t)(e0 * inv);
  o.h[1] = (half_t)(e1 * inv);
  *(uint*)(qout + g * 64 + lane * 2) = o.u;
}

// -------------- softmax over S=4096 for k, 8 dh-columns per block ----------
__global__ __launch_bounds__(256) void softmax_k_kernel(
    const half_t* __restrict__ kvlog,   // (B*S, 2048), k at col h*128+dh
    half_t* __restrict__ ks) {          // (B*S, 1024), col h*64+dh
  const int idx = blockIdx.x;           // b*128 + h*8 + g  (1024 blocks)
  const int g = idx & 7, h = (idx >> 3) & 15, b = idx >> 7;
  const long ibase = (long)b * SQ * 2048 + h * 128 + g * 8;
  const long obase = (long)b * SQ * 1024 + h * 64 + g * 8;
  __shared__ float red[256][8];
  const int t = threadIdx.x;

  float mx[8];
#pragma unroll
  for (int j = 0; j < 8; ++j) mx[j] = -1e30f;
  for (int s = t; s < SQ; s += 256) {
    union { uint4 u; half_t h[8]; } r;
    r.u = *(const uint4*)(kvlog + ibase + (long)s * 2048);
#pragma unroll
    for (int j = 0; j < 8; ++j) mx[j] = fmaxf(mx[j], (float)r.h[j]);
  }
#pragma unroll
  for (int j = 0; j < 8; ++j) red[t][j] = mx[j];
  __syncthreads();
  for (int off = 128; off > 0; off >>= 1) {
    if (t < off)
#pragma unroll
      for (int j = 0; j < 8; ++j) red[t][j] = fmaxf(red[t][j], red[t + off][j]);
    __syncthreads();
  }
#pragma unroll
  for (int j = 0; j < 8; ++j) mx[j] = red[0][j];
  __syncthreads();

  float sm[8] = {0.f, 0.f, 0.f, 0.f, 0.f, 0.f, 0.f, 0.f};
  for (int s = t; s < SQ; s += 256) {
    union { uint4 u; half_t h[8]; } r;
    r.u = *(const uint4*)(kvlog + ibase + (long)s * 2048);
#pragma unroll
    for (int j = 0; j < 8; ++j) sm[j] += __expf((float)r.h[j] - mx[j]);
  }
#pragma unroll
  for (int j = 0; j < 8; ++j) red[t][j] = sm[j];
  __syncthreads();
  for (int off = 128; off > 0; off >>= 1) {
    if (t < off)
#pragma unroll
      for (int j = 0; j < 8; ++j) red[t][j] += red[t + off][j];
    __syncthreads();
  }
  float inv[8];
#pragma unroll
  for (int j = 0; j < 8; ++j) inv[j] = 1.0f / red[0][j];

  for (int s = t; s < SQ; s += 256) {
    union { uint4 u; half_t h[8]; } r, o;
    r.u = *(const uint4*)(kvlog + ibase + (long)s * 2048);
#pragma unroll
    for (int j = 0; j < 8; ++j)
      o.h[j] = (half_t)(__expf((float)r.h[j] - mx[j]) * inv[j]);
    *(uint4*)(ks + obase + (long)s * 1024) = o.u;
  }
}

// -------- context = k^T @ v per (b,h): (64x4096)@(4096x64) -> 64x64 --------
// sKt: A-style [dh][s]; sVc: B-style [e][s]. Double-buffered, perm packs.
__global__ __launch_bounds__(128) void context_kernel(
    const half_t* __restrict__ ks,     // (B*S, 1024) softmaxed k
    const half_t* __restrict__ kvlog,  // (B*S, 2048), v at col h*128+64
    half_t* __restrict__ ctx) {        // (B*H, 64, 64)
  constexpr int LDT = 32 + 8;
  __shared__ half_t sKt[2][64 * LDT];
  __shared__ half_t sVc[2][64 * LDT];
  const int bh = blockIdx.x, h = bh & 15, b = bh >> 4;
  const int tid = threadIdx.x, lane = tid & 31, wave = tid >> 5;
  const int rp = tid >> 3, c0 = (tid & 7) * 8;   // 16 s-pairs x 8 cols
  const long kbase = (long)b * SQ * 1024 + h * 64;
  const long vbase = (long)b * SQ * 2048 + h * 128 + 64;

  auto stage = [&](int buf, int s0) {
    const half_t* gk = ks + kbase + (long)(s0 + 2 * rp) * 1024 + c0;
    uint4 k0v = *(const uint4*)gk;
    uint4 k1v = *(const uint4*)(gk + 1024);
    const half_t* gv = kvlog + vbase + (long)(s0 + 2 * rp) * 2048 + c0;
    uint4 v0v = *(const uint4*)gv;
    uint4 v1v = *(const uint4*)(gv + 2048);
    const uint* kd0 = (const uint*)&k0v; const uint* kd1 = (const uint*)&k1v;
    const uint* vd0 = (const uint*)&v0v; const uint* vd1 = (const uint*)&v1v;
#pragma unroll
    for (int w = 0; w < 4; ++w) {
      *(uint*)(sKt[buf] + (c0 + 2 * w)     * LDT + 2 * rp) = perm_lo(kd0[w], kd1[w]);
      *(uint*)(sKt[buf] + (c0 + 2 * w + 1) * LDT + 2 * rp) = perm_hi(kd0[w], kd1[w]);
      *(uint*)(sVc[buf] + (c0 + 2 * w)     * LDT + 2 * rp) = perm_lo(vd0[w], vd1[w]);
      *(uint*)(sVc[buf] + (c0 + 2 * w + 1) * LDT + 2 * rp) = perm_hi(vd0[w], vd1[w]);
    }
  };

  v8f acc[4] = {};
  stage(0, 0);
  __syncthreads();
  const int nIter = SQ / 32;
  for (int it = 0; it < nIter; ++it) {
    const int cur = it & 1;
    if (it + 1 < nIter) stage(cur ^ 1, (it + 1) * 32);

    const v16h afrag = frag_contig(
        sKt[cur] + (wave * 16 + (lane & 15)) * LDT + ((lane >> 4) << 4));
#pragma unroll
    for (int nt = 0; nt < 4; ++nt) {
      const v16h bfrag = frag_contig(
          sVc[cur] + (nt * 16 + (lane & 15)) * LDT + ((lane >> 4) << 4));
      acc[nt] = __builtin_amdgcn_wmma_f32_16x16x32_f16(
          false, afrag, false, bfrag, (short)0, acc[nt], false, false);
    }
    __syncthreads();
  }
#pragma unroll
  for (int nt = 0; nt < 4; ++nt) {
    const int n = nt * 16 + (lane & 15);
#pragma unroll
    for (int r = 0; r < 8; ++r) {
      const int m = wave * 16 + r + ((lane >> 4) << 3);
      ctx[((long)bh * 64 + m) * 64 + n] = (half_t)acc[nt][r];
    }
  }
}

// -------------- out = q_soft @ ctx per (b,h): (64x64)@(64x64) --------------
__global__ __launch_bounds__(128) void apply_ctx_kernel(
    const half_t* __restrict__ qs,    // (B*S, 1024)
    const half_t* __restrict__ ctx,   // (B*H, 64, 64)
    half_t* __restrict__ outh) {      // (B*S, 1024)
  constexpr int LDQ = 64 + 8;
  __shared__ half_t sQ[64 * LDQ];     // row-major [s][k]
  __shared__ half_t sCc[64 * LDQ];    // column-major [n][k]
  const int bh = blockIdx.y, h = bh & 15, b = bh >> 4;
  const int s0 = blockIdx.x * 64;
  const int tid = threadIdx.x, lane = tid & 31, wave = tid >> 5;

  {  // sQ: 64x64 row-major, 32 halves per thread, async DMA
    const int qrow = tid >> 1, qcol = (tid & 1) * 32;
    const half_t* gq =
        qs + ((long)(b * SQ + s0 + qrow)) * 1024 + h * 64 + qcol;
#pragma unroll
    for (int j = 0; j < 4; ++j)
      copy16_g2l(sQ + qrow * LDQ + qcol + j * 8, gq + j * 8);
  }
  {  // sCc: scatter ctx column-major, k-pair perm packs
    const int rp = tid >> 2, c0 = (tid & 3) * 16;   // 32 k-pairs x 16 cols
    const half_t* gc = ctx + ((long)bh * 64 + 2 * rp) * 64 + c0;
    uint4 r0a = ((const uint4*)gc)[0], r0b = ((const uint4*)gc)[1];
    uint4 r1a = ((const uint4*)(gc + 64))[0], r1b = ((const uint4*)(gc + 64))[1];
    const uint* d0a = (const uint*)&r0a; const uint* d0b = (const uint*)&r0b;
    const uint* d1a = (const uint*)&r1a; const uint* d1b = (const uint*)&r1b;
#pragma unroll
    for (int w = 0; w < 4; ++w) {
      *(uint*)(sCc + (c0 + 2 * w)          * LDQ + 2 * rp) = perm_lo(d0a[w], d1a[w]);
      *(uint*)(sCc + (c0 + 2 * w + 1)      * LDQ + 2 * rp) = perm_hi(d0a[w], d1a[w]);
      *(uint*)(sCc + (c0 + 8 + 2 * w)      * LDQ + 2 * rp) = perm_lo(d0b[w], d1b[w]);
      *(uint*)(sCc + (c0 + 8 + 2 * w + 1)  * LDQ + 2 * rp) = perm_hi(d0b[w], d1b[w]);
    }
  }
  async_fence();
  __syncthreads();

  v8f acc[4] = {};
#pragma unroll
  for (int k0 = 0; k0 < 64; k0 += 32) {
    const v16h afrag = frag_contig(
        sQ + (wave * 16 + (lane & 15)) * LDQ + k0 + ((lane >> 4) << 4));
#pragma unroll
    for (int nt = 0; nt < 4; ++nt) {
      const v16h bfrag = frag_contig(
          sCc + (nt * 16 + (lane & 15)) * LDQ + k0 + ((lane >> 4) << 4));
      acc[nt] = __builtin_amdgcn_wmma_f32_16x16x32_f16(
          false, afrag, false, bfrag, (short)0, acc[nt], false, false);
    }
  }
#pragma unroll
  for (int nt = 0; nt < 4; ++nt) {
    const int n = nt * 16 + (lane & 15);
#pragma unroll
    for (int r = 0; r < 8; ++r) {
      const int m = wave * 16 + r + ((lane >> 4) << 3);
      outh[(long)(b * SQ + s0 + m) * 1024 + h * 64 + n] = (half_t)acc[nt][r];
    }
  }
}

// ------------------------------- host side ---------------------------------
extern "C" void kernel_launch(void* const* d_in, const int* in_sizes, int n_in,
                              void* d_out, int out_size, void* d_ws, size_t ws_size,
                              hipStream_t stream) {
  const float* x    = (const float*)d_in[0];
  const float* Wq   = (const float*)d_in[1];
  const float* Wkv  = (const float*)d_in[2];
  const float* Wlin = (const float*)d_in[3];
  const float* blin = (const float*)d_in[4];
  float* out = (float*)d_out;

  half_t* ws = (half_t*)d_ws;
  size_t off = 0;
  auto alloc = [&](size_t n) { half_t* p = ws + off; off += n; return p; };
  half_t* xh    = alloc((size_t)BSQ * DQ);
  half_t* Wqh   = alloc((size_t)DQ * DMQ);
  half_t* Wkvh  = alloc((size_t)DQ * 2 * DMQ);
  half_t* Wlinh = alloc((size_t)DMQ * DMQ);
  half_t* qlog  = alloc((size_t)BSQ * DMQ);         // softmax in place
  half_t* kvlog = alloc((size_t)BSQ * 2 * DMQ);
  half_t* ks    = alloc((size_t)BSQ * DMQ);
  half_t* ctx   = alloc((size_t)BQ * HQ * DHQ * DHQ);
  half_t* outh  = kvlog;   // kv-logits dead after context_kernel

  const dim3 blk128(128), blk256(256);

  // 1) f32 -> f16 converts (8 elems/thread)
  {
    long n = (long)BSQ * DQ;
    f32_to_f16_kernel<<<(unsigned)(n / 8 / 256), blk256, 0, stream>>>(x, xh, n);
    n = (long)DQ * DMQ;
    f32_to_f16_kernel<<<(unsigned)(n / 8 / 256), blk256, 0, stream>>>(Wq, Wqh, n);
    n = (long)DQ * 2 * DMQ;
    f32_to_f16_kernel<<<(unsigned)(n / 8 / 256), blk256, 0, stream>>>(Wkv, Wkvh, n);
    n = (long)DMQ * DMQ;
    f32_to_f16_kernel<<<(unsigned)(n / 8 / 256), blk256, 0, stream>>>(Wlin, Wlinh, n);
  }

  // 2) q-logits = x @ Wq ; kv-logits = x @ Wkv   (block tile 128x64)
  gemm_f16_kernel<<<dim3(DMQ / 64, BSQ / 128), blk128, 0, stream>>>(
      xh, Wqh, nullptr, qlog, nullptr, BSQ, DMQ, DQ);
  gemm_f16_kernel<<<dim3(2 * DMQ / 64, BSQ / 128), blk128, 0, stream>>>(
      xh, Wkvh, nullptr, kvlog, nullptr, BSQ, 2 * DMQ, DQ);

  // 3) softmax(q over dh) * DH^-0.5, in place
  softmax_q_kernel<<<(unsigned)((long)BSQ * HQ / 8), blk256, 0, stream>>>(
      qlog, qlog, 0.125f);

  // 4) softmax(k over S), vectorized 8 columns/block
  softmax_k_kernel<<<BQ * HQ * 8, blk256, 0, stream>>>(kvlog, ks);

  // 5) context = k^T @ v per (b,h)
  context_kernel<<<BQ * HQ, blk128, 0, stream>>>(ks, kvlog, ctx);

  // 6) out_h = q_soft @ context per (b,h)
  apply_ctx_kernel<<<dim3(SQ / 64, BQ * HQ), blk128, 0, stream>>>(qlog, ctx, outh);

  // 7) final = out_h @ Wlin + blin -> f32 d_out
  gemm_f16_kernel<<<dim3(DMQ / 64, BSQ / 128), blk128, 0, stream>>>(
      outh, Wlinh, blin, nullptr, out, BSQ, DMQ, DQ);
}
